// AttentionOnlyBlock_89361089561463
// MI455X (gfx1250) — matrix operand order repaired and checked by reading, not compiled, not run on hardware
//
#include <hip/hip_runtime.h>
#include <hip/hip_bf16.h>
#include <math.h>

// ---------------------------------------------------------------- types
typedef __attribute__((ext_vector_type(16))) __bf16  v16bf;
typedef __attribute__((ext_vector_type(8)))  float   v8f;
typedef __attribute__((ext_vector_type(4)))  unsigned int u32x4;

#define Bb      2
#define Ss      2048
#define Dd      1024
#define Hh      16
#define HD      64
#define HIDDEN  2730
#define WIN     512
#define BS      (Bb*Ss)          // 4096 rows

static __device__ __forceinline__ __bf16 to_bf16(float f) { return (__bf16)f; }

// ---------------------------------------------------------------- f32 -> bf16 convert
__global__ __launch_bounds__(256) void k_cvt_bf16(const float* __restrict__ in,
                                                  __bf16* __restrict__ out, int n) {
  int i = blockIdx.x * 256 + threadIdx.x;
  if (i < n) out[i] = to_bf16(in[i]);
}

// ---------------------------------------------------------------- RMSNorm (row length D=1024) -> bf16
__global__ __launch_bounds__(256) void k_rmsnorm_bf16(const float* __restrict__ x,
                                                      const float* __restrict__ w,
                                                      __bf16* __restrict__ out) {
  const int row = blockIdx.x;
  const float* xr = x + (size_t)row * Dd;
  __shared__ float red[256];
  float ss = 0.f;
  for (int c = threadIdx.x; c < Dd; c += 256) { float v = xr[c]; ss += v * v; }
  red[threadIdx.x] = ss; __syncthreads();
  for (int s = 128; s > 0; s >>= 1) {
    if (threadIdx.x < s) red[threadIdx.x] += red[threadIdx.x + s];
    __syncthreads();
  }
  float inv = rsqrtf(red[0] * (1.f / Dd) + 1e-6f);
  for (int c = threadIdx.x; c < Dd; c += 256)
    out[(size_t)row * Dd + c] = to_bf16(xr[c] * inv * w[c]);
}

// ---------------------------------------------------------------- tiled bf16 WMMA GEMM
// C[M,N] (f32) = A[M,K](bf16, row-major) * W[N,K](bf16, row-major)^T  [+ addsrc]
// Relied on by fast paths: M % BM == 0, K even (all call sites satisfy).
#define BM 128
#define BN 64
#define BK 32
#define LPAD 16   // keeps 16B alignment of fragment chunks (row stride 96B)

// fragment build + 4 WMMAs for one staged K-tile
static __device__ __forceinline__ void gemm_tile_mma(
    const __bf16 (*sA)[BK + LPAD], const __bf16 (*sW)[BK + LPAD],
    int lane, int wm, int wn, v8f c[2][2]) {
  const int lrow  = lane & 15;
  const int khalf = lane >> 4;      // 0/1
  union Frag { u32x4 u[2]; v16bf v; };
  Frag a[2], b[2];
  #pragma unroll
  for (int i = 0; i < 2; ++i) {
    const int r   = wm * 32 + i * 16 + lrow;
    const int klo = khalf * 8;
    a[i].u[0] = *(const u32x4*)(&sA[r][klo]);       // K = klo..klo+7
    a[i].u[1] = *(const u32x4*)(&sA[r][16 + klo]);  // K = 16+klo..16+klo+7
  }
  #pragma unroll
  for (int j = 0; j < 2; ++j) {
    const int col = wn * 32 + j * 16 + lrow;
    const int kb  = khalf * 8;                       // K = kb..kb+7, 16+kb..16+kb+7
    b[j].u[0] = *(const u32x4*)(&sW[col][kb]);
    b[j].u[1] = *(const u32x4*)(&sW[col][16 + kb]);
  }
  #pragma unroll
  for (int i = 0; i < 2; ++i)
    #pragma unroll
    for (int j = 0; j < 2; ++j)
      c[i][j] = __builtin_amdgcn_wmma_f32_16x16x32_bf16(
          false, a[i].v, false, b[j].v, (short)0, c[i][j], false, false);
}

__global__ __launch_bounds__(256) void k_gemm_bf16(const __bf16* __restrict__ A,
                                                   const __bf16* __restrict__ W,
                                                   const float*  __restrict__ addsrc,
                                                   float* __restrict__ C,
                                                   int M, int N, int K) {
  __shared__ alignas(16) __bf16 sA[BM][BK + LPAD];
  __shared__ alignas(16) __bf16 sW[BN][BK + LPAD];

  const int tid  = threadIdx.x;
  const int lane = tid & 31;
  const int wid  = tid >> 5;          // 0..7
  const int wm   = wid & 3;           // 4 waves along M (32 rows each)
  const int wn   = wid >> 2;          // 2 waves along N (32 cols each)
  const int bm0  = blockIdx.y * BM;
  const int bn0  = blockIdx.x * BN;

  v8f c[2][2];
  for (int i = 0; i < 2; ++i)
    for (int j = 0; j < 2; ++j)
      for (int e = 0; e < 8; ++e) c[i][j][e] = 0.f;

  const int ar = tid >> 1, ac = (tid & 1) * 16;   // A-tile: 16 bf16/thread
  const int wr = tid >> 2, wc = (tid & 3) * 8;    // W-tile: 8 bf16/thread
  const int gw = bn0 + wr;
  // clamp: rows >= N stage garbage that only feeds cols >= N (never stored)
  const int gwc = (gw < N) ? gw : (N - 1);
  const size_t arowbase = (size_t)(bm0 + ar) * K; // row always valid (M % BM == 0)
  const size_t wrowbase = (size_t)gwc * K;

  const int kfull = K & ~(BK - 1);

  // ---- main loop: branch-free vector staging (dword aligned: K even)
  for (int k0 = 0; k0 < kfull; k0 += BK) {
    {
      const unsigned int* srcA = (const unsigned int*)(A + arowbase + k0 + ac);
      unsigned int* dstA = (unsigned int*)(&sA[ar][ac]);
      #pragma unroll
      for (int u = 0; u < 8; ++u) dstA[u] = srcA[u];

      const unsigned int* srcW = (const unsigned int*)(W + wrowbase + k0 + wc);
      unsigned int* dstW = (unsigned int*)(&sW[wr][wc]);
      #pragma unroll
      for (int u = 0; u < 4; ++u) dstW[u] = srcW[u];

      __builtin_prefetch(A + arowbase + k0 + ac + BK, 0, 1);
    }
    __syncthreads();
    gemm_tile_mma(sA, sW, lane, wm, wn, c);
    __syncthreads();
  }

  // ---- K tail (executes once, only when K % BK != 0): zero-padded stage
  if (kfull < K) {
    #pragma unroll
    for (int u = 0; u < 16; ++u) {
      int gc = kfull + ac + u;
      sA[ar][ac + u] = (gc < K) ? A[arowbase + gc] : to_bf16(0.f);
    }
    #pragma unroll
    for (int u = 0; u < 8; ++u) {
      int gc = kfull + wc + u;
      sW[wr][wc + u] = (gc < K) ? W[wrowbase + gc] : to_bf16(0.f);
    }
    __syncthreads();
    gemm_tile_mma(sA, sW, lane, wm, wn, c);
  }

  // ---- store: vgpr e -> row e + 8*(lane>>4), col = lane&15
  #pragma unroll
  for (int i = 0; i < 2; ++i)
    #pragma unroll
    for (int j = 0; j < 2; ++j) {
      const int col = bn0 + wn * 32 + j * 16 + (lane & 15);
      if (col < N) {
        #pragma unroll
        for (int e = 0; e < 8; ++e) {
          const int r = bm0 + wm * 32 + i * 16 + e + 8 * (lane >> 4);
          float v = c[i][j][e];
          if (addsrc) v += addsrc[(size_t)r * N + col];
          C[(size_t)r * N + col] = v;
        }
      }
    }
}

// ---------------------------------------------------------------- head split + q/k RMSNorm + RoPE -> bf16 [B,H,S,HD]
__global__ __launch_bounds__(64) void k_qkv_heads(const float* __restrict__ qkv,
                                                  const float* __restrict__ qw,
                                                  const float* __restrict__ kw,
                                                  __bf16* __restrict__ qo,
                                                  __bf16* __restrict__ ko,
                                                  __bf16* __restrict__ vo) {
  const int idx = blockIdx.x;
  const int h  = idx & (Hh - 1);
  const int bs = idx >> 4;            // b*S + s
  const int b  = bs >> 11;
  const int s  = bs & (Ss - 1);
  const int d  = threadIdx.x;

  const float* r = qkv + (size_t)bs * (3 * Dd);
  float q = r[h * HD + d];
  float k = r[Dd + h * HD + d];
  float v = r[2 * Dd + h * HD + d];

  __shared__ float red[64], qb[64], kb[64];
  red[d] = q * q; __syncthreads();
  for (int st = 32; st > 0; st >>= 1) { if (d < st) red[d] += red[d + st]; __syncthreads(); }
  float qi = rsqrtf(red[0] * (1.f / HD) + 1e-6f);
  __syncthreads();
  red[d] = k * k; __syncthreads();
  for (int st = 32; st > 0; st >>= 1) { if (d < st) red[d] += red[d + st]; __syncthreads(); }
  float ki = rsqrtf(red[0] * (1.f / HD) + 1e-6f);
  __syncthreads();

  float qn = q * qi * qw[d];
  float kn = k * ki * kw[d];
  qb[d] = qn; kb[d] = kn; __syncthreads();

  const int j = d & 31;
  float inv_freq = __expf(-((float)(2 * j) / (float)HD) * 9.210340371976184f); // ln 1e4
  float ang = (float)s * inv_freq;
  float cs = __cosf(ang), sn = __sinf(ang);
  float qrp = (d < 32) ? -qb[d + 32] : qb[d - 32];
  float krp = (d < 32) ? -kb[d + 32] : kb[d - 32];
  float qr = qn * cs + qrp * sn;
  float kr = kn * cs + krp * sn;

  const size_t o = ((size_t)(b * Hh + h) * Ss + s) * HD + d;
  qo[o] = to_bf16(qr * 0.125f);   // fold HD^-0.5 into q
  ko[o] = to_bf16(kr);
  vo[o] = to_bf16(v);
}

// ---------------------------------------------------------------- flash attention, sliding window, WMMA
// grid.x = (S/64) * B*H ; block = 128 (4 waves, 16 q-rows each)
__global__ __launch_bounds__(128) void k_attn(const __bf16* __restrict__ qg,
                                              const __bf16* __restrict__ kg,
                                              const __bf16* __restrict__ vg,
                                              __bf16* __restrict__ ctx) {
  const int lane = threadIdx.x & 31;
  const int w    = threadIdx.x >> 5;            // wave id 0..3
  const int qblk = blockIdx.x % (Ss / 64);
  const int bh   = blockIdx.x / (Ss / 64);
  const int b    = bh >> 4, h = bh & 15;
  const int q0   = qblk * 64 + w * 16;

  const __bf16* qp = qg + (size_t)bh * Ss * HD;
  const __bf16* kp = kg + (size_t)bh * Ss * HD;
  const __bf16* vp = vg + (size_t)bh * Ss * HD;

  __shared__ alignas(16) float  s_sc [4][16][36];   // per-wave 16x32 scores/probs
  __shared__ alignas(16) __bf16 s_v  [4][32][72];   // per-wave V tile 32x64 (+pad)
  __shared__ float s_max[4][16], s_sum[4][16], s_fac[4][16];

  union FragA { u32x4 u[2]; v16bf v; };

  // Q fragments (d 0..31 and d 32..63)
  FragA aq0, aq1;
  {
    const int r   = q0 + (lane & 15);
    const int klo = (lane >> 4) * 16;
    const __bf16* base = qp + (size_t)r * HD;
    aq0.u[0] = *(const u32x4*)(base + klo);
    aq0.u[1] = *(const u32x4*)(base + klo + 8);
    aq1.u[0] = *(const u32x4*)(base + 32 + klo);
    aq1.u[1] = *(const u32x4*)(base + 32 + klo + 8);
  }

  v8f acc[4];
  for (int t = 0; t < 4; ++t)
    for (int e = 0; e < 8; ++e) acc[t][e] = 0.f;

  if (lane < 16) { s_max[w][lane] = -1e30f; s_sum[w][lane] = 0.f; s_fac[w][lane] = 1.f; }
  asm volatile("s_wait_dscnt 0" ::: "memory");

  int kbeg = q0 - (WIN - 1); if (kbeg < 0) kbeg = 0; kbeg &= ~31;
  const int kend = q0 + 16;

  for (int kt = kbeg; kt < kend; kt += 32) {
    // ---- stage V tile (32 rows x 64) into LDS: lane copies one row
    {
      int rr = kt + lane; if (rr > Ss - 1) rr = Ss - 1;
      const u32x4* src = (const u32x4*)(vp + (size_t)rr * HD);
      u32x4* dst = (u32x4*)(&s_v[w][lane][0]);
      #pragma unroll
      for (int cchunk = 0; cchunk < 8; ++cchunk) dst[cchunk] = src[cchunk];
    }

    // ---- scores for two 16-col tiles
    #pragma unroll
    for (int j = 0; j < 2; ++j) {
      int kc = kt + j * 16 + (lane & 15); if (kc > Ss - 1) kc = Ss - 1;
      const int db = (lane >> 4) * 16;
      const __bf16* kbase = kp + (size_t)kc * HD;
      FragA bk0, bk1;
      bk0.u[0] = *(const u32x4*)(kbase + db);
      bk0.u[1] = *(const u32x4*)(kbase + db + 8);
      bk1.u[0] = *(const u32x4*)(kbase + 32 + db);
      bk1.u[1] = *(const u32x4*)(kbase + 32 + db + 8);

      v8f s;
      for (int e = 0; e < 8; ++e) s[e] = 0.f;
      s = __builtin_amdgcn_wmma_f32_16x16x32_bf16(false, aq0.v, false, bk0.v,
                                                  (short)0, s, false, false);
      s = __builtin_amdgcn_wmma_f32_16x16x32_bf16(false, aq1.v, false, bk1.v,
                                                  (short)0, s, false, false);

      const int gcol = kt + j * 16 + (lane & 15);
      #pragma unroll
      for (int e = 0; e < 8; ++e) {
        const int ml   = e + 8 * (lane >> 4);
        const int grow = q0 + ml;
        float val = s[e];
        if (gcol > grow || gcol < grow - (WIN - 1)) val = -1e30f;
        s_sc[w][ml][j * 16 + (lane & 15)] = val;
      }
    }
    asm volatile("s_wait_dscnt 0" ::: "memory");

    // ---- online softmax update (lanes 0..15, one row each)
    if (lane < 16) {
      float m32 = -1e30f;
      #pragma unroll
      for (int cidx = 0; cidx < 32; ++cidx) {
        float sv = s_sc[w][lane][cidx];
        m32 = fmaxf(m32, sv);
      }
      const float old = s_max[w][lane];
      const float nm  = fmaxf(old, m32);
      const float fac = __expf(old - nm);
      float psum = 0.f;
      #pragma unroll
      for (int cidx = 0; cidx < 32; ++cidx) {
        float sv = s_sc[w][lane][cidx];
        float pv = (sv <= -1e29f) ? 0.f : __expf(sv - nm);
        s_sc[w][lane][cidx] = pv;
        psum += pv;
      }
      s_sum[w][lane] = s_sum[w][lane] * fac + psum;
      s_max[w][lane] = nm;
      s_fac[w][lane] = fac;
    }
    asm volatile("s_wait_dscnt 0" ::: "memory");

    // ---- rescale accumulators by per-row factor
    #pragma unroll
    for (int t = 0; t < 4; ++t)
      #pragma unroll
      for (int e = 0; e < 8; ++e)
        acc[t][e] *= s_fac[w][e + 8 * (lane >> 4)];

    // ---- P (bf16 A-frag from LDS) x V (B-frag from LDS)
    FragA ap;
    {
      const int r   = lane & 15;
      const int klo = (lane >> 4) * 16;
      #pragma unroll
      for (int t = 0; t < 8; ++t) {
        ap.v[t]     = to_bf16(s_sc[w][r][klo + t]);
        ap.v[8 + t] = to_bf16(s_sc[w][r][klo + 8 + t]);
      }
    }
    #pragma unroll
    for (int t = 0; t < 4; ++t) {
      FragA bv;
      const int dcol = t * 16 + (lane & 15);
      const int kb2  = (lane >> 4) * 16;
      #pragma unroll
      for (int u = 0; u < 16; ++u) bv.v[u] = s_v[w][kb2 + u][dcol];
      acc[t] = __builtin_amdgcn_wmma_f32_16x16x32_bf16(false, ap.v, false, bv.v,
                                                       (short)0, acc[t], false, false);
    }
  }

  asm volatile("s_wait_dscnt 0" ::: "memory");

  // ---- epilogue: normalize and scatter to ctx[b, s, h*64+d] (bf16)
  #pragma unroll
  for (int e = 0; e < 8; ++e) {
    const int ml   = e + 8 * (lane >> 4);
    const int grow = q0 + ml;
    const float inv = 1.f / s_sum[w][ml];
    const size_t base = ((size_t)(b * Ss + grow)) * Dd + h * HD;
    #pragma unroll
    for (int t = 0; t < 4; ++t) {
      const int dcol = t * 16 + (lane & 15);
      ctx[base + dcol] = to_bf16(acc[t][e] * inv);
    }
  }
}

// ---------------------------------------------------------------- silu(t1)*t3 -> bf16
__global__ __launch_bounds__(256) void k_silumul(const float* __restrict__ t1,
                                                 const float* __restrict__ t3,
                                                 __bf16* __restrict__ g, int n) {
  int i = blockIdx.x * 256 + threadIdx.x;
  if (i < n) {
    float a = t1[i];
    g[i] = to_bf16((a / (1.f + __expf(-a))) * t3[i]);
  }
}

// ---------------------------------------------------------------- host
extern "C" void kernel_launch(void* const* d_in, const int* in_sizes, int n_in,
                              void* d_out, int out_size, void* d_ws, size_t ws_size,
                              hipStream_t stream) {
  (void)in_sizes; (void)n_in; (void)out_size; (void)ws_size;
  const float* x    = (const float*)d_in[0];
  const float* n1w  = (const float*)d_in[1];
  const float* n2w  = (const float*)d_in[2];
  const float* wqkv = (const float*)d_in[3];
  const float* qnw  = (const float*)d_in[4];
  const float* knw  = (const float*)d_in[5];
  const float* wo   = (const float*)d_in[6];
  const float* w1   = (const float*)d_in[7];
  const float* w3   = (const float*)d_in[8];
  const float* w2   = (const float*)d_in[9];
  float* out = (float*)d_out;

  char* p = (char*)d_ws;
  auto alloc = [&](size_t bytes) -> char* {
    char* r = p; p += (bytes + 255) & ~(size_t)255; return r;
  };

  __bf16* wqkv_bf = (__bf16*)alloc((size_t)3 * Dd * Dd * 2);
  __bf16* wo_bf   = (__bf16*)alloc((size_t)Dd * Dd * 2);
  __bf16* w1_bf   = (__bf16*)alloc((size_t)HIDDEN * Dd * 2);
  __bf16* w3_bf   = (__bf16*)alloc((size_t)HIDDEN * Dd * 2);
  __bf16* w2_bf   = (__bf16*)alloc((size_t)HIDDEN * Dd * 2);
  __bf16* h_bf    = (__bf16*)alloc((size_t)BS * Dd * 2);   // reused for h2
  __bf16* q_bf    = (__bf16*)alloc((size_t)BS * Dd * 2);
  __bf16* k_bf    = (__bf16*)alloc((size_t)BS * Dd * 2);
  __bf16* v_bf    = (__bf16*)alloc((size_t)BS * Dd * 2);
  __bf16* ctx_bf  = (__bf16*)alloc((size_t)BS * Dd * 2);
  float*  xmid    = (float*)alloc((size_t)BS * Dd * 4);
  __bf16* g_bf    = (__bf16*)alloc((size_t)BS * HIDDEN * 2);
  // big region: qkv f32 (phase 1) reused by t1/t3 (phase 2)
  char* big = alloc((size_t)BS * HIDDEN * 4 * 2);
  float* qkv_f = (float*)big;
  float* t1_f  = (float*)big;
  float* t3_f  = (float*)(big + (size_t)BS * HIDDEN * 4);

  const int nqkv = 3 * Dd * Dd, no = Dd * Dd, nh = HIDDEN * Dd;
  k_cvt_bf16<<<(nqkv + 255) / 256, 256, 0, stream>>>(wqkv, wqkv_bf, nqkv);
  k_cvt_bf16<<<(no   + 255) / 256, 256, 0, stream>>>(wo,   wo_bf,   no);
  k_cvt_bf16<<<(nh   + 255) / 256, 256, 0, stream>>>(w1,   w1_bf,   nh);
  k_cvt_bf16<<<(nh   + 255) / 256, 256, 0, stream>>>(w3,   w3_bf,   nh);
  k_cvt_bf16<<<(nh   + 255) / 256, 256, 0, stream>>>(w2,   w2_bf,   nh);

  // h = rmsnorm(x) (bf16)
  k_rmsnorm_bf16<<<BS, 256, 0, stream>>>(x, n1w, h_bf);

  // qkv = h @ w_qkv^T
  k_gemm_bf16<<<dim3((3 * Dd + BN - 1) / BN, (BS + BM - 1) / BM), 256, 0, stream>>>(
      h_bf, wqkv_bf, nullptr, qkv_f, BS, 3 * Dd, Dd);

  // heads + q/k norm + rope
  k_qkv_heads<<<BS * Hh, 64, 0, stream>>>(qkv_f, qnw, knw, q_bf, k_bf, v_bf);

  // attention
  k_attn<<<(Ss / 64) * Bb * Hh, 128, 0, stream>>>(q_bf, k_bf, v_bf, ctx_bf);

  // xmid = x + ctx @ w_o^T
  k_gemm_bf16<<<dim3(Dd / BN, BS / BM), 256, 0, stream>>>(
      ctx_bf, wo_bf, x, xmid, BS, Dd, Dd);

  // h2 = rmsnorm(xmid)
  k_rmsnorm_bf16<<<BS, 256, 0, stream>>>(xmid, n2w, h_bf);

  // t1 = h2 @ w1^T ; t3 = h2 @ w3^T
  k_gemm_bf16<<<dim3((HIDDEN + BN - 1) / BN, BS / BM), 256, 0, stream>>>(
      h_bf, w1_bf, nullptr, t1_f, BS, HIDDEN, Dd);
  k_gemm_bf16<<<dim3((HIDDEN + BN - 1) / BN, BS / BM), 256, 0, stream>>>(
      h_bf, w3_bf, nullptr, t3_f, BS, HIDDEN, Dd);

  // g = silu(t1)*t3 (bf16)
  const int ng = BS * HIDDEN;
  k_silumul<<<(ng + 255) / 256, 256, 0, stream>>>(t1_f, t3_f, g_bf, ng);

  // out = xmid + g @ w2^T
  k_gemm_bf16<<<dim3(Dd / BN, BS / BM), 256, 0, stream>>>(
      g_bf, w2_bf, xmid, out, BS, Dd, HIDDEN);
}
